// SinkhornWarpInterpolator_4131758539064
// MI455X (gfx1250) — matrix operand order repaired and checked by reading, not compiled, not run on hardware
//
#include <hip/hip_runtime.h>
#include <hip/hip_bf16.h>
#include <math.h>

typedef __attribute__((ext_vector_type(16))) _Float16 v16h;
typedef __attribute__((ext_vector_type(8)))  _Float16 v8h;
typedef __attribute__((ext_vector_type(8)))  float    v8f;
typedef __attribute__((ext_vector_type(4)))  float    v4f;

#define NTOK 1024   // 32x32 tokens
#define DIM  256    // 16 ch * 4 * 4 patch
#define NTR  405    // 5 angles * 9 * 9 shifts

// ---------------------------------------------------------------------------
// Bilinear SE(2) taps for token-grid position (i=row/y, j=col/x), 32x32 grid,
// normalized coords in [-1,1], zero-padding via zero weights (matches
// _grid_sample_zeros). Tap order matches reference: (y0,x0),(y0,x1),(y1,x0),(y1,x1).
// ---------------------------------------------------------------------------
__device__ __forceinline__ void se2_taps(float c, float s, float tx, float ty,
                                         int i, int j, int* toks, float* w) {
  float X = -1.0f + (2.0f / 31.0f) * (float)j;
  float Y = -1.0f + (2.0f / 31.0f) * (float)i;
  float gx = c * X - s * Y + tx;
  float gy = s * X + c * Y + ty;
  float fx = (gx + 1.0f) * 15.5f;
  float fy = (gy + 1.0f) * 15.5f;
  float x0 = floorf(fx), y0 = floorf(fy);
  float wx = fx - x0, wy = fy - y0;
  float ys2[2] = {y0, y0 + 1.0f};
  float xs2[2] = {x0, x0 + 1.0f};
  float wyv[2] = {1.0f - wy, wy};
  float wxv[2] = {1.0f - wx, wx};
#pragma unroll
  for (int a = 0; a < 2; ++a) {
#pragma unroll
    for (int b = 0; b < 2; ++b) {
      int idx = a * 2 + b;
      float yi = ys2[a], xi = xs2[b];
      bool valid = (yi >= 0.0f) && (yi <= 31.0f) && (xi >= 0.0f) && (xi <= 31.0f);
      int yc = (int)fminf(fmaxf(yi, 0.0f), 31.0f);
      int xc = (int)fminf(fmaxf(xi, 0.0f), 31.0f);
      toks[idx] = yc * 32 + xc;
      w[idx] = valid ? (wyv[a] * wxv[b]) : 0.0f;
    }
  }
}

// ---------------------------------------------------------------------------
// Kernel 1: patch tokens -> L2-normalized f16 features.
// One wave (32 lanes) per token; lane handles 8 of 256 dims:
// dim d = c*16 + py*4 + px ; lane -> c = lane>>1, py0 = (lane&1)*2.
// ---------------------------------------------------------------------------
__global__ void feat_kernel(const float* __restrict__ z0, const float* __restrict__ z1,
                            _Float16* __restrict__ F0h, _Float16* __restrict__ F1h) {
  int wid  = threadIdx.x >> 5;
  int lane = threadIdx.x & 31;
  int g    = blockIdx.x * 8 + wid;     // 0..2047
  int img  = g >> 10;
  int tok  = g & 1023;
  int ti   = tok >> 5, tj = tok & 31;

  const float* z = img ? z1 : z0;
  int c   = lane >> 1;
  int py0 = (lane & 1) * 2;
  const float* p0 = z + c * (128 * 128) + (ti * 4 + py0) * 128 + tj * 4;
  v4f r0 = *(const v4f*)p0;
  v4f r1 = *(const v4f*)(p0 + 128);

  float ss = r0.x * r0.x + r0.y * r0.y + r0.z * r0.z + r0.w * r0.w
           + r1.x * r1.x + r1.y * r1.y + r1.z * r1.z + r1.w * r1.w;
#pragma unroll
  for (int off = 16; off > 0; off >>= 1)
    ss += __shfl_xor(ss, off, 32);
  float inv = 1.0f / fmaxf(sqrtf(ss), 1e-6f);

  _Float16* dst = (img ? F1h : F0h) + tok * DIM + lane * 8;
  v8h o;
  o[0] = (_Float16)(r0.x * inv); o[1] = (_Float16)(r0.y * inv);
  o[2] = (_Float16)(r0.z * inv); o[3] = (_Float16)(r0.w * inv);
  o[4] = (_Float16)(r1.x * inv); o[5] = (_Float16)(r1.y * inv);
  o[6] = (_Float16)(r1.z * inv); o[7] = (_Float16)(r1.w * inv);
  *(v8h*)dst = o;
}

// ---------------------------------------------------------------------------
// Kernel 2: K = F0 * F1^T  (1024x1024, k=256) via v_wmma_f32_16x16x32_f16.
// One wave per 16x16 output tile; 8 WMMAs along k.
// A lane (m=r, half h): a[0..7]=A[m][kb+ h*8 .. +7], a[8..15]=A[m][kb+16+h*8 ..]
// B lane (n=r, half h): b[0..7]=B[kb+h*8 .. +7][n], b[8..15]=B[kb+16+h*8 ..][n] (rows of F1)
// C lane: c[i] -> row mb*16 + h*8 + i, col nb*16 + r.
// ---------------------------------------------------------------------------
__global__ void gemm_kernel(const _Float16* __restrict__ F0h,
                            const _Float16* __restrict__ F1h,
                            float* __restrict__ Kmat) {
  int wid  = threadIdx.x >> 5;
  int lane = threadIdx.x & 31;
  int t    = blockIdx.x * 8 + wid;     // 0..4095 tiles
  int mb   = t >> 6, nb = t & 63;
  int hlf  = lane >> 4;
  int r    = lane & 15;

  const _Float16* arow = F0h + (mb * 16 + r) * DIM;
  const _Float16* brow = F1h + (nb * 16 + r) * DIM;

  v8f acc = {};
#pragma unroll
  for (int kb = 0; kb < 8; ++kb) {
    int kbase = kb * 32;
    v8h a0 = *(const v8h*)(arow + kbase + hlf * 8);
    v8h a1 = *(const v8h*)(arow + kbase + 16 + hlf * 8);
    v8h b0 = *(const v8h*)(brow + kbase + hlf * 8);
    v8h b1 = *(const v8h*)(brow + kbase + 16 + hlf * 8);
    v16h av, bv;
#pragma unroll
    for (int q = 0; q < 8; ++q) {
      av[q] = a0[q]; av[q + 8] = a1[q];
      bv[q] = b0[q]; bv[q + 8] = b1[q];
    }
    acc = __builtin_amdgcn_wmma_f32_16x16x32_f16(false, av, false, bv,
                                                 (short)0, acc, false, false);
  }

  int col  = nb * 16 + r;
  int row0 = mb * 16 + hlf * 8;
#pragma unroll
  for (int q = 0; q < 8; ++q)
    Kmat[(row0 + q) * NTOK + col] = acc[q];
}

// ---------------------------------------------------------------------------
// Kernel 3: score[t] = sum_pos sum_taps w * K[pos][tap]  (argmax-equivalent
// to mean(f0m * warp(f1m)) since the mean divisor is constant).
// One block per transform; 256 threads, 4 positions each; LDS tree reduce.
// ---------------------------------------------------------------------------
__global__ void score_kernel(const float* __restrict__ Kmat, float* __restrict__ scores) {
  int t   = blockIdx.x;              // 0..404
  int ai  = t / 81, rem = t % 81;
  int dyi = rem / 9, dxi = rem % 9;
  const float deg2rad = 0.017453292519943295f;
  float theta = (float)(ai * 5 - 10) * deg2rad;
  float dx = (float)(dxi - 4), dy = (float)(dyi - 4);
  float tx = 2.0f * dx / 31.0f, ty = 2.0f * dy / 31.0f;
  float c = cosf(theta), s = sinf(theta);

  float acc = 0.0f;
  for (int pos = threadIdx.x; pos < NTOK; pos += 256) {
    int i = pos >> 5, j = pos & 31;
    int toks[4]; float w[4];
    se2_taps(c, s, tx, ty, i, j, toks, w);
    const float* Krow = Kmat + pos * NTOK;
    acc += w[0] * Krow[toks[0]] + w[1] * Krow[toks[1]]
         + w[2] * Krow[toks[2]] + w[3] * Krow[toks[3]];
  }

  __shared__ float red[256];
  red[threadIdx.x] = acc;
  __syncthreads();
#pragma unroll
  for (int off = 128; off > 0; off >>= 1) {
    if (threadIdx.x < off) red[threadIdx.x] += red[threadIdx.x + off];
    __syncthreads();
  }
  if (threadIdx.x == 0) scores[t] = red[0];
}

// ---------------------------------------------------------------------------
// Kernel 4: argmax over 405 scores (first-max tie-break, like jnp.argmax);
// writes best transform params {theta, tx, ty, dx, dy}.
// ---------------------------------------------------------------------------
__global__ void argmax_kernel(const float* __restrict__ scores, float* __restrict__ best) {
  int lane = threadIdx.x;  // 32 threads, one wave
  float bv = -1e30f; int bi = 0x7fffffff;
  for (int t = lane; t < NTR; t += 32) {
    float v = scores[t];
    if (v > bv || (v == bv && t < bi)) { bv = v; bi = t; }
  }
#pragma unroll
  for (int off = 16; off > 0; off >>= 1) {
    float ov = __shfl_xor(bv, off, 32);
    int   oi = __shfl_xor(bi, off, 32);
    if (ov > bv || (ov == bv && oi < bi)) { bv = ov; bi = oi; }
  }
  if (lane == 0) {
    int ai = bi / 81, rem = bi % 81;
    int dyi = rem / 9, dxi = rem % 9;
    const float deg2rad = 0.017453292519943295f;
    float dx = (float)(dxi - 4), dy = (float)(dyi - 4);
    best[0] = (float)(ai * 5 - 10) * deg2rad;
    best[1] = 2.0f * dx / 31.0f;
    best[2] = 2.0f * dy / 31.0f;
    best[3] = dx;
    best[4] = dy;
  }
}

// ---------------------------------------------------------------------------
// Kernel 5: per-window Sinkhorn (65x65, 20 iters) in LDS + flow composition.
// Logits for (n, m) = 1.25 * sum_taps(m) w * K[tok0(n)][tap], dustbin = -2.
// One block (256 thr) per window; 16 windows.
// ---------------------------------------------------------------------------
__global__ void sinkhorn_kernel(const float* __restrict__ Kmat,
                                const float* __restrict__ best,
                                float* __restrict__ out) {
  __shared__ float lp[65][66];       // padded stride
  __shared__ int   tapTok[64][4];
  __shared__ float tapW[64][4];
  __shared__ float sb[8];

  int tid = threadIdx.x;
  int wwy = blockIdx.x >> 2;
  int wwx = blockIdx.x & 3;

  if (tid < 5) sb[tid] = best[tid];
  __syncthreads();
  float theta = sb[0], tx = sb[1], ty = sb[2], bdx = sb[3], bdy = sb[4];
  float cth = cosf(theta), sth = sinf(theta);

  if (tid < 64) {
    int my = tid >> 3, mx = tid & 7;
    int toks[4]; float w[4];
    se2_taps(cth, sth, tx, ty, wwy * 8 + my, wwx * 8 + mx, toks, w);
#pragma unroll
    for (int q = 0; q < 4; ++q) { tapTok[tid][q] = toks[q]; tapW[tid][q] = w[q]; }
  }
  __syncthreads();

  // Fill log-coupling matrix
  for (int e = tid; e < 65 * 65; e += 256) {
    int n = e / 65, m = e % 65;
    float v = -2.0f;                 // DUSTBIN
    if (n < 64 && m < 64) {
      int ny = n >> 3, nx = n & 7;
      int tok0 = (wwy * 8 + ny) * 32 + (wwx * 8 + nx);
      const float* Krow = Kmat + tok0 * NTOK;
      float d = tapW[m][0] * Krow[tapTok[m][0]] + tapW[m][1] * Krow[tapTok[m][1]]
              + tapW[m][2] * Krow[tapTok[m][2]] + tapW[m][3] * Krow[tapTok[m][3]];
      v = 1.25f * d;                 // 1/(sqrt(256)*0.05)
    }
    lp[n][m] = v;
  }
  __syncthreads();

  for (int it = 0; it < 20; ++it) {
    if (tid < 65) {                  // row normalize (axis 2)
      float mx = -1e30f;
      for (int m = 0; m < 65; ++m) mx = fmaxf(mx, lp[tid][m]);
      float sum = 0.0f;
      for (int m = 0; m < 65; ++m) sum += expf(lp[tid][m] - mx);
      float lse = mx + logf(sum);
      for (int m = 0; m < 65; ++m) lp[tid][m] -= lse;
    }
    __syncthreads();
    if (tid < 65) {                  // col normalize (axis 1)
      float mx = -1e30f;
      for (int n = 0; n < 65; ++n) mx = fmaxf(mx, lp[n][tid]);
      float sum = 0.0f;
      for (int n = 0; n < 65; ++n) sum += expf(lp[n][tid] - mx);
      float lse = mx + logf(sum);
      for (int n = 0; n < 65; ++n) lp[n][tid] -= lse;
    }
    __syncthreads();
  }

  if (tid < 64) {
    int ny = tid >> 3, nx = tid & 7;
    float mass = 0.0f, qx = 0.0f, qy = 0.0f;
    for (int m = 0; m < 64; ++m) {
      float p = expf(lp[tid][m]);
      mass += p;
      qx += p * (float)(m & 7);
      qy += p * (float)(m >> 3);
    }
    mass = fmaxf(mass, 1e-8f);
    float ddx = qx / mass - (float)nx;
    float ddy = qy / mass - (float)ny;
    float conf = 1.0f - expf(lp[tid][64]);

    int gi = wwy * 8 + ny, gj = wwx * 8 + nx;
    float vx = (float)gj - 15.5f + ddx;
    float vy = (float)gi - 15.5f + ddy;
    float qgx = cth * vx - sth * vy + 15.5f + bdx;
    float qgy = sth * vx + cth * vy + 15.5f + bdy;
    int o = gi * 32 + gj;
    out[o * 2 + 0]   = qgx - (float)gj;
    out[o * 2 + 1]   = qgy - (float)gi;
    out[2048 + o]    = conf;
  }
}

// ---------------------------------------------------------------------------
extern "C" void kernel_launch(void* const* d_in, const int* in_sizes, int n_in,
                              void* d_out, int out_size, void* d_ws, size_t ws_size,
                              hipStream_t stream) {
  const float* z0 = (const float*)d_in[0];
  const float* z1 = (const float*)d_in[1];
  char* ws = (char*)d_ws;

  _Float16* F0h   = (_Float16*)(ws);                     // 512 KB
  _Float16* F1h   = (_Float16*)(ws + 524288);            // 512 KB
  float*    Kmat  = (float*)   (ws + 1048576);           // 4 MB
  float*    scores= (float*)   (ws + 5242880);           // 405 floats
  float*    best  = (float*)   (ws + 5242880 + 4096);    // 5 floats
  float*    out   = (float*)d_out;

  feat_kernel   <<<256, 256, 0, stream>>>(z0, z1, F0h, F1h);
  gemm_kernel   <<<512, 256, 0, stream>>>(F0h, F1h, Kmat);
  score_kernel  <<<NTR, 256, 0, stream>>>(Kmat, scores);
  argmax_kernel <<<1,   32,  0, stream>>>(scores, best);
  sinkhorn_kernel<<<16, 256, 0, stream>>>(Kmat, best, out);
}
